// MMDLoss_62294205661835
// MI455X (gfx1250) — matrix operand order repaired and checked by reading, not compiled, not run on hardware
//
#include <hip/hip_runtime.h>

// ---------------------------------------------------------------------------
// MMD loss, MI455X (gfx1250, wave32, WMMA + async-to-LDS) — fused
// Gram / exp / binned reduction.
//   pass 0: fp32 -> bf16 hi/lo split + row norms + mask ids
//   pass 1: bandwidth via column sums (O(N*D), no NxN matrix needed)
//   pass 2: 3x (4096^2 x 1024) Gram via v_wmma_f32_16x16x32_bf16 (hi/lo x3),
//           double-buffered global_load_async_to_lds_b128 pipeline,
//           symmetric tt/ss blocks computed once (upper triangle x2),
//           fused exp + masked bin accumulation (10 scalar bins total)
//   pass 3: finalize scalar loss
// ---------------------------------------------------------------------------

typedef __attribute__((ext_vector_type(16))) __bf16 v16bf;
typedef __attribute__((ext_vector_type(8)))  float  v8f;
typedef unsigned int u32;
typedef unsigned short u16;
typedef unsigned long long u64;

#define NPTS 4096
#define DIMF 1024
#define EPS_F 1e-12f
#define TSTRIDE 20          // uints per LDS tile row: 16 data + 4 pad
                            //  -> 80B rows: 16B aligned (async b128 stores),
                            //     20*r mod 64 distinct for r=0..15 (banks)
#define TILE_U32 (128 * TSTRIDE)        // uints per tile
#define TILE_B   (TILE_U32 * 4)         // bytes per tile

static __device__ __forceinline__ u16 f2bf(float f) {
  u32 u = __builtin_bit_cast(u32, f);
  u32 r = u + 0x7fffu + ((u >> 16) & 1u);   // round-to-nearest-even
  return (u16)(r >> 16);
}
static __device__ __forceinline__ float bf2f(u16 h) {
  u32 u = ((u32)h) << 16;
  return __builtin_bit_cast(float, u);
}

struct U8 { uint2 x, y, z, w; };

// Gather one 16x32 bf16 WMMA operand fragment from an LDS tile row pointer.
// o0/o1 select the two 16-byte K-pair groups; callers use contiguous k = 16h + s.
static __device__ __forceinline__ v16bf ld_frag(const u32* rowp, int o0, int o1) {
  U8 u;
  u.x = *(const uint2*)(rowp + o0);
  u.y = *(const uint2*)(rowp + o0 + 2);
  u.z = *(const uint2*)(rowp + o1);
  u.w = *(const uint2*)(rowp + o1 + 2);
  return __builtin_bit_cast(v16bf, u);
}

// CDNA5 async copy: 16 bytes global -> LDS, tracked by ASYNCcnt.
static __device__ __forceinline__ void async_b128(u32 lds_byte_off, u64 gaddr) {
  asm volatile("global_load_async_to_lds_b128 %0, %1, off"
               :: "v"(lds_byte_off), "v"(gaddr) : "memory");
}
static __device__ __forceinline__ void wait_async0() {
  asm volatile("s_wait_asynccnt 0x0" ::: "memory");
}

// ---------------------------------------------------------------------------
// Pass 0a: bf16 hi/lo split + row norms.  One block per (matrix,row).
// ---------------------------------------------------------------------------
__global__ void __launch_bounds__(256) k_prep(
    const float* __restrict__ T, const float* __restrict__ S,
    u16* __restrict__ Thi, u16* __restrict__ Tlo,
    u16* __restrict__ Shi, u16* __restrict__ Slo,
    float* __restrict__ nT, float* __restrict__ nS) {
  __shared__ float red[256];
  const int bid = blockIdx.x;
  const int mat = bid >> 12;
  const int row = bid & (NPTS - 1);
  const float* src = (mat ? S : T) + (size_t)row * DIMF;
  u16* dh = (mat ? Shi : Thi) + (size_t)row * DIMF;
  u16* dl = (mat ? Slo : Tlo) + (size_t)row * DIMF;
  float acc = 0.f;
#pragma unroll
  for (int e = 0; e < 4; ++e) {
    int idx = threadIdx.x + e * 256;
    float f = src[idx];
    u16 h = f2bf(f);
    float hf = bf2f(h);
    u16 l = f2bf(f - hf);
    dh[idx] = h;
    dl[idx] = l;
    acc += f * f;
  }
  red[threadIdx.x] = acc;
  __syncthreads();
  for (int s = 128; s > 0; s >>= 1) {
    if (threadIdx.x < s) red[threadIdx.x] += red[threadIdx.x + s];
    __syncthreads();
  }
  if (threadIdx.x == 0) (mat ? nS : nT)[row] = red[0];
}

// ---------------------------------------------------------------------------
// Pass 0b: mask ids.  tb[i] = subgroups[i,0]; sg[i] = 2*sub[i,0] + sub[i,1].
// ---------------------------------------------------------------------------
__global__ void k_mask(const int* __restrict__ sub,
                       int* __restrict__ tb, int* __restrict__ sg) {
  int i = blockIdx.x * 256 + threadIdx.x;
  if (i < NPTS) {
    int a0 = (sub[i * 2 + 0] != 0);
    int a1 = (sub[i * 2 + 1] != 0);
    tb[i] = a0;
    sg[i] = a0 * 2 + a1;
  }
}

// ---------------------------------------------------------------------------
// Pass 1a: column sums of T and S (for the analytic mean-distance).
// ---------------------------------------------------------------------------
__global__ void k_colsum(const float* __restrict__ T, const float* __restrict__ S,
                         float* __restrict__ cT, float* __restrict__ cS) {
  int d = blockIdx.x * 256 + threadIdx.x;
  if (d >= DIMF) return;
  float st = 0.f, ss = 0.f;
  for (int i = 0; i < NPTS; ++i) {
    st += T[(size_t)i * DIMF + d];
    ss += S[(size_t)i * DIMF + d];
  }
  cT[d] = st;
  cS[d] = ss;
}

// ---------------------------------------------------------------------------
// Pass 1b: bandwidth scalar  inv = 1/(2*sqrt(mean(d_ts))), zero the bins.
// mean(d_ts) = ( N*(sum nT + sum nS) - 2 * dot(colsumT, colsumS) ) / N^2
// ---------------------------------------------------------------------------
__global__ void k_scalar(const float* __restrict__ nT, const float* __restrict__ nS,
                         const float* __restrict__ cT, const float* __restrict__ cS,
                         float* __restrict__ scal, float* __restrict__ gbins) {
  __shared__ float red[256];
  float a = 0.f;
  for (int i = threadIdx.x; i < NPTS; i += 256) a += nT[i] + nS[i];
  red[threadIdx.x] = a;
  __syncthreads();
  for (int s = 128; s > 0; s >>= 1) {
    if (threadIdx.x < s) red[threadIdx.x] += red[threadIdx.x + s];
    __syncthreads();
  }
  float sumN = red[0];
  __syncthreads();
  float b = 0.f;
  for (int d = threadIdx.x; d < DIMF; d += 256) b += cT[d] * cS[d];
  red[threadIdx.x] = b;
  __syncthreads();
  for (int s = 128; s > 0; s >>= 1) {
    if (threadIdx.x < s) red[threadIdx.x] += red[threadIdx.x + s];
    __syncthreads();
  }
  if (threadIdx.x == 0) {
    float dotc = red[0];
    float meanD = ((float)NPTS * sumN - 2.f * dotc) / ((float)NPTS * (float)NPTS);
    meanD = fmaxf(meanD, EPS_F);
    scal[0] = 1.f / (2.f * __builtin_sqrtf(meanD));
  }
  if (threadIdx.x < 16) gbins[threadIdx.x] = 0.f;
}

// ---------------------------------------------------------------------------
// Pass 2: fused Gram(WMMA bf16 hi/lo x3) -> exp -> masked bin reduction.
// grid = (32, 32, 3);  z: 0 = tt, 1 = ss, 2 = ts.   Block tile 128x128.
// 8 waves: wave (wr,wc) owns 32 rows x 64 cols = 2x4 sixteen-tiles.
// Double-buffered async global->LDS pipeline; symmetric passes skip by<bx.
// ---------------------------------------------------------------------------
__global__ void __launch_bounds__(256) k_mmd(
    const u16* __restrict__ Thi, const u16* __restrict__ Tlo,
    const u16* __restrict__ Shi, const u16* __restrict__ Slo,
    const float* __restrict__ nT, const float* __restrict__ nS,
    const int* __restrict__ tb, const int* __restrict__ sg,
    const float* __restrict__ scal, float* __restrict__ gbins) {
  __shared__ u32 tiles[2 * 4 * TILE_U32];   // [buf][Xh,Xl,Yh,Yl][128][TSTRIDE]
  __shared__ float sb[4];

  const int p = blockIdx.z;
  // symmetric passes (tt, ss): compute upper-triangle blocks only, weight 2
  if (p != 2 && blockIdx.y < blockIdx.x) return;

  const u16 *Xh, *Xl, *Yh, *Yl;
  const float *nx, *ny;
  if (p == 0)      { Xh = Thi; Xl = Tlo; Yh = Thi; Yl = Tlo; nx = nT; ny = nT; }
  else if (p == 1) { Xh = Shi; Xl = Slo; Yh = Shi; Yl = Slo; nx = nS; ny = nS; }
  else             { Xh = Thi; Xl = Tlo; Yh = Shi; Yl = Slo; nx = nT; ny = nS; }

  const int I0 = blockIdx.x * 128;
  const int J0 = blockIdx.y * 128;
  const int tid = threadIdx.x;
  const int lane = tid & 31;
  const int wv = tid >> 5;
  const int wr = wv & 3;          // row group (x32)
  const int wc = wv >> 2;         // col group (x64)
  const int l16 = (lane >= 16);
  const int ln = lane & 15;
  const float inv = scal[0];
  const float wgt = (p != 2 && blockIdx.y > blockIdx.x) ? 2.f : 1.f;

  v8f acc[2][4] = {};

  // cooperative loader mapping: thread -> (row lr, K-half lh)
  const int lr = tid & 127;
  const int lh = tid >> 7;
  const u64 gX[2] = { (u64)(uintptr_t)Xh + 2ull * (u64)(I0 + lr) * DIMF,
                      (u64)(uintptr_t)Xl + 2ull * (u64)(I0 + lr) * DIMF };
  const u64 gY[2] = { (u64)(uintptr_t)Yh + 2ull * (u64)(J0 + lr) * DIMF,
                      (u64)(uintptr_t)Yl + 2ull * (u64)(J0 + lr) * DIMF };
  const u32 ldsBase = (u32)(uintptr_t)(&tiles[0]);
  const u32 ldsRow = 80u * (u32)lr + 32u * (u32)lh;   // bytes within a tile

  const int b0 = l16 ? 8 : 0;     // A-fragment K-pair group offset (uints)
  const int c0 = l16 ? 8 : 0;     // B-fragment K-pair group offset (uints)

  // issue one K-chunk (32 wide) of all four tiles into buffer `buf`
  auto issue_tile = [&](int k0, int buf) {
    const u64 koff = 2ull * (u64)k0 + 32ull * (u64)lh;
#pragma unroll
    for (int a = 0; a < 4; ++a) {
      const u64 g = (a < 2 ? gX[a] : gY[a - 2]) + koff;
      const u32 l = ldsBase + (u32)(buf * 4 + a) * TILE_B + ldsRow;
      async_b128(l, g);
      async_b128(l + 16u, g + 16ull);
    }
  };

  issue_tile(0, 0);

  for (int kk = 0; kk < DIMF / 32; ++kk) {
    const int cur = kk & 1;
    wait_async0();        // own async writes for tile kk complete
    __syncthreads();      // everyone's writes done; prev reads of other buf done
    if (kk + 1 < DIMF / 32) issue_tile((kk + 1) * 32, 1 - cur);

    const u32* tXh = &tiles[(cur * 4 + 0) * TILE_U32];
    const u32* tXl = &tiles[(cur * 4 + 1) * TILE_U32];
    const u32* tYh = &tiles[(cur * 4 + 2) * TILE_U32];
    const u32* tYl = &tiles[(cur * 4 + 3) * TILE_U32];

    v16bf aH[2], aL[2];
#pragma unroll
    for (int ti = 0; ti < 2; ++ti) {
      int arow = wr * 32 + ti * 16 + ln;
      aH[ti] = ld_frag(&tXh[arow * TSTRIDE], b0, b0 + 4);
      aL[ti] = ld_frag(&tXl[arow * TSTRIDE], b0, b0 + 4);
    }
#pragma unroll
    for (int tj = 0; tj < 4; ++tj) {
      int brow = wc * 64 + tj * 16 + ln;
      v16bf bH = ld_frag(&tYh[brow * TSTRIDE], c0, c0 + 4);
      v16bf bL = ld_frag(&tYl[brow * TSTRIDE], c0, c0 + 4);
#pragma unroll
      for (int ti = 0; ti < 2; ++ti) {
        acc[ti][tj] = __builtin_amdgcn_wmma_f32_16x16x32_bf16(
            false, aH[ti], false, bH, (short)0, acc[ti][tj], false, false);
        acc[ti][tj] = __builtin_amdgcn_wmma_f32_16x16x32_bf16(
            false, aH[ti], false, bL, (short)0, acc[ti][tj], false, false);
        acc[ti][tj] = __builtin_amdgcn_wmma_f32_16x16x32_bf16(
            false, aL[ti], false, bH, (short)0, acc[ti][tj], false, false);
      }
    }
  }

  // epilogue: d = nx + ny - 2G, K = exp(-d*inv), accumulate into mask bins
  float loc[4] = {0.f, 0.f, 0.f, 0.f};
#pragma unroll
  for (int tj = 0; tj < 4; ++tj) {
    int jg = J0 + wc * 64 + tj * 16 + ln;
    float nyj = ny[jg];
    int mj = (p == 0) ? tb[jg] : sg[jg];
#pragma unroll
    for (int ti = 0; ti < 2; ++ti) {
#pragma unroll
      for (int r = 0; r < 8; ++r) {
        int ig = I0 + wr * 32 + ti * 16 + r + (l16 ? 8 : 0);
        float g = acc[ti][tj][r];
        float d = nx[ig] + nyj - 2.f * g;
        d = fmaxf(d, EPS_F);
        float kv = wgt * __expf(-d * inv);
        int mi = (p == 1) ? sg[ig] : tb[ig];
        if (p == 2) {
          if (mi == (mj >> 1)) loc[mj] += kv;
        } else {
          if (mi == mj) loc[mi] += kv;
        }
      }
    }
  }

  if (tid < 4) sb[tid] = 0.f;
  __syncthreads();
  const int nb = (p == 0) ? 2 : 4;
  for (int b = 0; b < nb; ++b)
    if (loc[b] != 0.f) atomicAdd(&sb[b], loc[b]);
  __syncthreads();
  const int base = (p == 0) ? 0 : ((p == 1) ? 2 : 6);
  if (tid < nb) atomicAdd(&gbins[base + tid], sb[tid]);
}

// ---------------------------------------------------------------------------
// Pass 3: counts + final scalar combination.
// ---------------------------------------------------------------------------
__global__ void k_final(const float* __restrict__ gbins, const int* __restrict__ tb,
                        const int* __restrict__ sg, float* __restrict__ out) {
  __shared__ int cnt[6];
  if (threadIdx.x < 6) cnt[threadIdx.x] = 0;
  __syncthreads();
  for (int i = threadIdx.x; i < NPTS; i += 256) {
    atomicAdd(&cnt[tb[i]], 1);
    atomicAdd(&cnt[2 + sg[i]], 1);
  }
  __syncthreads();
  if (threadIdx.x == 0) {
    float total = 0.f;
    for (int g = 0; g < 4; ++g) {
      int c0 = g >> 1;
      float tc = (float)cnt[c0];
      float sc = (float)cnt[2 + g];
      float per = gbins[c0]     / fmaxf(tc * tc, 1.f)
                + gbins[2 + g]  / fmaxf(sc * sc, 1.f)
                - 2.f * gbins[6 + g] / fmaxf(tc * sc, 1.f);
      if (sc > 0.f) total += per;
    }
    out[0] = total;
  }
}

// ---------------------------------------------------------------------------
extern "C" void kernel_launch(void* const* d_in, const int* in_sizes, int n_in,
                              void* d_out, int out_size, void* d_ws, size_t ws_size,
                              hipStream_t stream) {
  (void)in_sizes; (void)n_in; (void)out_size; (void)ws_size;
  const float* T = (const float*)d_in[0];
  const float* S = (const float*)d_in[1];
  const int* sub = (const int*)d_in[2];

  char* ws = (char*)d_ws;
  size_t off = 0;
  auto carve = [&](size_t bytes) -> void* {
    void* pp = ws + off;
    off += (bytes + 255) & ~(size_t)255;
    return pp;
  };
  const size_t feat16 = (size_t)NPTS * DIMF * sizeof(u16);
  u16* Thi = (u16*)carve(feat16);
  u16* Tlo = (u16*)carve(feat16);
  u16* Shi = (u16*)carve(feat16);
  u16* Slo = (u16*)carve(feat16);
  float* nT = (float*)carve(NPTS * sizeof(float));
  float* nS = (float*)carve(NPTS * sizeof(float));
  float* cT = (float*)carve(DIMF * sizeof(float));
  float* cS = (float*)carve(DIMF * sizeof(float));
  int* tb   = (int*)carve(NPTS * sizeof(int));
  int* sg   = (int*)carve(NPTS * sizeof(int));
  float* scal  = (float*)carve(64 * sizeof(float));
  float* gbins = (float*)carve(64 * sizeof(float));

  k_prep<<<2 * NPTS, 256, 0, stream>>>(T, S, Thi, Tlo, Shi, Slo, nT, nS);
  k_mask<<<(NPTS + 255) / 256, 256, 0, stream>>>(sub, tb, sg);
  k_colsum<<<DIMF / 256, 256, 0, stream>>>(T, S, cT, cS);
  k_scalar<<<1, 256, 0, stream>>>(nT, nS, cT, cS, scal, gbins);
  k_mmd<<<dim3(NPTS / 128, NPTS / 128, 3), 256, 0, stream>>>(
      Thi, Tlo, Shi, Slo, nT, nS, tb, sg, scal, gbins);
  k_final<<<1, 256, 0, stream>>>(gbins, tb, sg, (float*)d_out);
}
